// Model_489626272140
// MI455X (gfx1250) — matrix operand order repaired and checked by reading, not compiled, not run on hardware
//
#include <hip/hip_runtime.h>
#include <hip/hip_bf16.h>

typedef __attribute__((ext_vector_type(16))) _Float16 v16h;
typedef __attribute__((ext_vector_type(8)))  float    v8f;

#define N_PTS    1000000
#define NUM_EMB  6572
#define EMB_DIM  32
#define HIDDEN   128
#define TILES    (N_PTS / 16)          /* 62500, exact */
#define WPB      8                     /* waves per block */
#define BLOCK    (WPB * 32)
#define GRID     2048
#define FROW     136                   /* padded staging row, halves (bank-conflict-free) */

#define WMMA_F16(a, b, c) \
  __builtin_amdgcn_wmma_f32_16x16x32_f16(false, (a), false, (b), (short)0, (c), false, false)

// B fragment: two contiguous 16B chunks per lane from pre-swizzled LDS.
// fragBase points at [2][32][8] _Float16 = [half-group q][lane][8].
static __device__ __forceinline__ v16h load_b_frag(const _Float16* fragBase, int lane) {
  union { int4 q[2]; v16h h; } B;
  const int4* p = (const int4*)fragBase;
  B.q[0] = p[lane];        // q=0 : halves e=0..7
  B.q[1] = p[32 + lane];   // q=1 : halves e=8..15
  return B.h;
}

// A fragment (16x32 f16): lane holds row M=lane%16.
// Half e -> K = (e<8 ? e : e+8) + (lane>=16 ? 8 : 0); pairs are contiguous -> b32 loads.
static __device__ __forceinline__ v16h load_a_frag(const _Float16* row, int kbase, int hiOff) {
  union { unsigned u[8]; v16h h; } A;
#pragma unroll
  for (int d = 0; d < 8; ++d) {
    int K = kbase + 2 * d + ((d >= 4) ? 8 : 0) + hiOff;
    A.u[d] = *(const unsigned*)(row + K);
  }
  return A.h;
}

__launch_bounds__(BLOCK) __global__
void mlp_lerp_kernel(const float* __restrict__ x,
                     const float* __restrict__ emb,
                     const float* __restrict__ table,
                     const float* __restrict__ W0,
                     const float* __restrict__ b0,
                     const float* __restrict__ W1,
                     const float* __restrict__ b1,
                     const float* __restrict__ W2,
                     const float* __restrict__ b2,
                     float* __restrict__ out) {
  // Pre-swizzled B fragments: [k-chunk][n-tile][q][lane][8 halves]
  __shared__ __align__(16) _Float16 w0frag[2][8][2][32][8];   // 16 KB, K padded 34->64 with 0
  __shared__ __align__(16) _Float16 w1frag[4][8][2][32][8];   // 32 KB
  __shared__ __align__(16) _Float16 stag[WPB][16][FROW];      // 34 KB, feat0/feat1 overlaid
  __shared__ float b0s[HIDDEN], b1s[HIDDEN], w2s[HIDDEN];

  const int tid = threadIdx.x;

  // ---- one-time weight swizzle (f32 -> f16 B-fragment layout) ----
  for (int i = tid; i < 2 * 8 * 2 * 32 * 8; i += BLOCK) {
    int lane = i & 31;
    int d = (i >> 5) & 7, q = (i >> 8) & 1, j = (i >> 9) & 7, c = (i >> 12) & 1;
    int K   = 32 * c + ((lane & 16) ? 8 : 0) + q * 16 + d;
    int col = 16 * j + (lane & 15);
    float w = (K < 34) ? W0[K * HIDDEN + col] : 0.0f;
    w0frag[c][j][q][lane][d] = (_Float16)w;
  }
  for (int i = tid; i < 4 * 8 * 2 * 32 * 8; i += BLOCK) {
    int lane = i & 31;
    int d = (i >> 5) & 7, q = (i >> 8) & 1, j = (i >> 9) & 7, c = (i >> 12) & 3;
    int K   = 32 * c + ((lane & 16) ? 8 : 0) + q * 16 + d;
    int col = 16 * j + (lane & 15);
    w1frag[c][j][q][lane][d] = (_Float16)W1[K * HIDDEN + col];
  }
  for (int i = tid; i < HIDDEN; i += BLOCK) {
    b0s[i] = b0[i];
    b1s[i] = b1[i];
    w2s[i] = W2[i];
  }
  __syncthreads();

  const int wave  = tid >> 5;
  const int lane  = tid & 31;
  const int laneM = lane & 15;
  const int hiOff = (lane & 16) ? 8 : 0;
  _Float16* myRow = &stag[wave][laneM][0];

  for (int tile = blockIdx.x * WPB + wave; tile < TILES; tile += (int)gridDim.x * WPB) {
    const int pbase = tile << 4;

    // ---- stage input features (f16): [x0,x1, lerp-emb(32), zeros..63] ----
    {
      const int p   = pbase + laneM;
      const int part = (lane >> 4) & 1;          // lo half: feats 0..17, hi half: 18..33 + pad
      float ev  = emb[p];
      int   e1  = (int)ev;
      int   e2  = e1 + 1; if (e2 > NUM_EMB - 1) e2 = NUM_EMB - 1;
      float res = ev - (float)e1;
      const float4* r1 = (const float4*)(table + e1 * EMB_DIM + part * 16);
      const float4* r2 = (const float4*)(table + e2 * EMB_DIM + part * 16);
      _Float16* dst = &stag[wave][laneM][0];
      if (!part) {
        float2 xv = ((const float2*)x)[p];
        dst[0] = (_Float16)xv.x;
        dst[1] = (_Float16)xv.y;
      }
#pragma unroll
      for (int t = 0; t < 4; ++t) {
        float4 a = r1[t], b = r2[t];
        int f = 2 + part * 16 + t * 4;
        dst[f + 0] = (_Float16)(a.x + (b.x - a.x) * res);
        dst[f + 1] = (_Float16)(a.y + (b.y - a.y) * res);
        dst[f + 2] = (_Float16)(a.z + (b.z - a.z) * res);
        dst[f + 3] = (_Float16)(a.w + (b.w - a.w) * res);
      }
      if (part) {
#pragma unroll
        for (int z = 34; z < 72; z += 2) *(unsigned*)(dst + z) = 0u;  // zero K=34..63(+pad)
      }
    }
    __asm__ volatile("s_wait_dscnt 0" ::: "memory");

    // ---- load A fragments for layer 0 (16x64 over two K-chunks) ----
    v16h A0c0 = load_a_frag(myRow, 0, hiOff);
    v16h A0c1 = load_a_frag(myRow, 32, hiOff);

    // ---- layer 0: h1 = clamp01(featA @ W0 + b0), write f16 to staging ----
#pragma unroll
    for (int j = 0; j < 8; ++j) {
      v8f acc = {};
      acc = WMMA_F16(A0c0, load_b_frag(&w0frag[0][j][0][0][0], lane), acc);
      acc = WMMA_F16(A0c1, load_b_frag(&w0frag[1][j][0][0][0], lane), acc);
      const int col = 16 * j + laneM;
      const float bias = b0s[col];
#pragma unroll
      for (int r = 0; r < 8; ++r) {
        float v = acc[r] + bias;
        v = fminf(fmaxf(v, 0.0f), 1.0f);
        stag[wave][r + hiOff][col] = (_Float16)v;   // D(M=r+hiOff, N=col) -> feat row M
      }
    }
    __asm__ volatile("s_wait_dscnt 0" ::: "memory");

    // ---- layer 1: h2 = clamp01(h1 @ W1 + b1) ----
    v8f acc1[8];
#pragma unroll
    for (int j = 0; j < 8; ++j) { v8f z = {}; acc1[j] = z; }
#pragma unroll
    for (int c = 0; c < 4; ++c) {
      v16h Ac = load_a_frag(myRow, 32 * c, hiOff);
#pragma unroll
      for (int j = 0; j < 8; ++j)
        acc1[j] = WMMA_F16(Ac, load_b_frag(&w1frag[c][j][0][0][0], lane), acc1[j]);
    }

    // ---- layer 2 folded in: partial[r] = sum_col clamp01(h2)*W2[col] ----
    float partial[8];
#pragma unroll
    for (int r = 0; r < 8; ++r) partial[r] = 0.0f;
#pragma unroll
    for (int j = 0; j < 8; ++j) {
      const int col = 16 * j + laneM;
      const float bias = b1s[col];
      const float wv = w2s[col];
#pragma unroll
      for (int r = 0; r < 8; ++r) {
        float v = fminf(fmaxf(acc1[j][r] + bias, 0.0f), 1.0f);
        partial[r] += v * wv;
      }
    }
    // reduce across the 16 lanes of each half-wave (N dimension)
#pragma unroll
    for (int r = 0; r < 8; ++r) {
#pragma unroll
      for (int m = 1; m <= 8; m <<= 1) partial[r] += __shfl_xor(partial[r], m, 32);
    }
    if (laneM == 0) {
      const float bb = b2[0];
      float4 o0, o1;
      o0.x = 1.0f / (1.0f + __expf(-(partial[0] + bb)));
      o0.y = 1.0f / (1.0f + __expf(-(partial[1] + bb)));
      o0.z = 1.0f / (1.0f + __expf(-(partial[2] + bb)));
      o0.w = 1.0f / (1.0f + __expf(-(partial[3] + bb)));
      o1.x = 1.0f / (1.0f + __expf(-(partial[4] + bb)));
      o1.y = 1.0f / (1.0f + __expf(-(partial[5] + bb)));
      o1.z = 1.0f / (1.0f + __expf(-(partial[6] + bb)));
      o1.w = 1.0f / (1.0f + __expf(-(partial[7] + bb)));
      float* op = out + pbase + hiOff;       // lane 0 -> pts 0..7, lane 16 -> pts 8..15
      *(float4*)op       = o0;
      *((float4*)op + 1) = o1;
    }
  }
}

extern "C" void kernel_launch(void* const* d_in, const int* in_sizes, int n_in,
                              void* d_out, int out_size, void* d_ws, size_t ws_size,
                              hipStream_t stream) {
  (void)in_sizes; (void)n_in; (void)out_size; (void)d_ws; (void)ws_size;
  const float* x   = (const float*)d_in[0];
  const float* emb = (const float*)d_in[1];
  const float* tab = (const float*)d_in[2];
  const float* W0  = (const float*)d_in[3];
  const float* b0  = (const float*)d_in[4];
  const float* W1  = (const float*)d_in[5];
  const float* b1  = (const float*)d_in[6];
  const float* W2  = (const float*)d_in[7];
  const float* b2  = (const float*)d_in[8];
  float* out = (float*)d_out;
  mlp_lerp_kernel<<<dim3(GRID), dim3(BLOCK), 0, stream>>>(
      x, emb, tab, W0, b0, W1, b1, W2, b2, out);
}
